// LSTM_POS_Tagger_48455821034129
// MI455X (gfx1250) — compile-verified, matrix operand order checked by reading, not run on hardware
//
#include <hip/hip_runtime.h>
#include <math.h>
#include <stdint.h>

// ---------------------------------------------------------------- types
typedef __attribute__((ext_vector_type(2)))  float  v2f;
typedef __attribute__((ext_vector_type(8)))  float  v8f;
typedef __attribute__((ext_vector_type(8)))  __bf16 v8bf;
typedef __attribute__((ext_vector_type(16))) __bf16 v16bf;

// ---------------------------------------------------------------- sizes
#define S_  8192
#define C_  24
#define E_  512
#define H_  1024
#define R_  256
#define L_  50
#define EPR (E_ + R_)      // 768
#define G4H 4096           // 4*H
#define NBLK_W 64          // blocks cooperating on the word recurrence

__device__ __forceinline__ float sigf(float x) { return 1.0f / (1.0f + expf(-x)); }

// ---------------------------------------------------------------- init: zero h0 + sync counter (graph-replay safe)
__global__ void k_init(float* hbuf, unsigned int* cnt) {
    int t = threadIdx.x;
    if (t < H_) hbuf[t] = 0.0f;
    if (t == 0) *cnt = 0u;
}

// ---------------------------------------------------------------- f32 -> bf16 weight copies
__global__ void k_f32_to_bf16(const float* __restrict__ src, __bf16* __restrict__ dst, int n) {
    int i = blockIdx.x * blockDim.x + threadIdx.x;
    if (i < n) dst[i] = (__bf16)src[i];
}

// ---------------------------------------------------------------- embedding gather into X[:, 0:512] (bf16)
__global__ void k_embed(const int* __restrict__ sent, const float* __restrict__ emb,
                        __bf16* __restrict__ Xbf) {
    int s = blockIdx.x;
    long long row = sent[s];
    const float* src = emb + row * (long long)E_;
    __bf16* dst = Xbf + (size_t)s * EPR;
    for (int k = threadIdx.x; k < E_; k += blockDim.x) dst[k] = (__bf16)src[k];
}

// ---------------------------------------------------------------- char LSTM: 16 words/block, f32 WMMA 16x16x4
// gates[M=words 16, N=4R] = x*Wc_ih + bc + h @ Wc_hh^T ; h kept in LDS, c in D-layout registers.
#define HSTR 260   // LDS row stride (floats), padded to break bank conflicts
__global__ __launch_bounds__(256) void k_char_lstm(
    const float* __restrict__ chars, const float* __restrict__ Wih,
    const float* __restrict__ Whh,   const float* __restrict__ bc,
    __bf16* __restrict__ Xbf)
{
    __shared__ float hl[16 * HSTR];
    __shared__ float wih_s[4 * R_];
    __shared__ float b_s[4 * R_];
    __shared__ float xw[16];

    const int tid  = threadIdx.x;
    const int lane = tid & 31;
    const int wave = tid >> 5;
    const int m    = lane & 15;      // A row / D col-within-tile
    const int hi   = lane >> 4;      // half-wave select
    const int kb   = hi * 2;         // f32 fragment K offset {0,2}
    const int wbase = blockIdx.x * 16;
    const int nc   = m;              // column within 16-wide N tile

    for (int i = tid; i < 4 * R_; i += 256) { wih_s[i] = Wih[i]; b_s[i] = bc[i]; }
    for (int i = tid; i < 16 * HSTR; i += 256) hl[i] = 0.0f;

    v8f cst[2];
    #pragma unroll
    for (int u = 0; u < 2; ++u)
        #pragma unroll
        for (int r = 0; r < 8; ++r) cst[u][r] = 0.0f;

    #pragma unroll 1
    for (int t = 0; t < C_; ++t) {
        if (tid < 16) xw[tid] = chars[(size_t)(wbase + tid) * C_ + t];
        __syncthreads();

        v8f acc[2][4];
        #pragma unroll
        for (int u = 0; u < 2; ++u) {
            const int ut = wave + u * 8;      // this wave's two N tiles (of 16)
            const int n0 = ut * 16;
            // accumulator init: x * Wc_ih + bc  (rank-1 input term + bias)
            #pragma unroll
            for (int g = 0; g < 4; ++g) {
                const float wg = wih_s[g * R_ + n0 + nc];
                const float bg = b_s[g * R_ + n0 + nc];
                #pragma unroll
                for (int r = 0; r < 8; ++r)
                    acc[u][g][r] = xw[r + hi * 8] * wg + bg;
            }
            const float* bp0 = Whh + (size_t)(0 * R_ + n0 + nc) * R_ + kb;
            const float* bp1 = Whh + (size_t)(1 * R_ + n0 + nc) * R_ + kb;
            const float* bp2 = Whh + (size_t)(2 * R_ + n0 + nc) * R_ + kb;
            const float* bp3 = Whh + (size_t)(3 * R_ + n0 + nc) * R_ + kb;
            const float* ap  = &hl[m * HSTR + kb];
            #pragma unroll 4
            for (int kk = 0; kk < R_; kk += 4) {
                v2f a = *(const v2f*)(ap + kk);
                v2f b0 = *(const v2f*)(bp0 + kk);
                v2f b1 = *(const v2f*)(bp1 + kk);
                v2f b2 = *(const v2f*)(bp2 + kk);
                v2f b3 = *(const v2f*)(bp3 + kk);
                acc[u][0] = __builtin_amdgcn_wmma_f32_16x16x4_f32(false, a, false, b0, (short)0, acc[u][0], false, false);
                acc[u][1] = __builtin_amdgcn_wmma_f32_16x16x4_f32(false, a, false, b1, (short)0, acc[u][1], false, false);
                acc[u][2] = __builtin_amdgcn_wmma_f32_16x16x4_f32(false, a, false, b2, (short)0, acc[u][2], false, false);
                acc[u][3] = __builtin_amdgcn_wmma_f32_16x16x4_f32(false, a, false, b3, (short)0, acc[u][3], false, false);
            }
        }
        __syncthreads();   // all reads of hl done -> safe to overwrite
        #pragma unroll
        for (int u = 0; u < 2; ++u) {
            const int n0 = (wave + u * 8) * 16;
            #pragma unroll
            for (int r = 0; r < 8; ++r) {
                float iv = acc[u][0][r], fv = acc[u][1][r];
                float gv = acc[u][2][r], ov = acc[u][3][r];
                float cn = sigf(fv) * cst[u][r] + sigf(iv) * tanhf(gv);
                float hn = sigf(ov) * tanhf(cn);
                cst[u][r] = cn;
                hl[(r + hi * 8) * HSTR + n0 + nc] = hn;
            }
        }
        // next-iteration __syncthreads orders these writes before reads
    }
    __syncthreads();
    // final h is the char representation -> X[:, 512:768] (bf16)
    for (int i = tid; i < 16 * R_; i += 256) {
        int mw = i >> 8, n = i & (R_ - 1);
        Xbf[(size_t)(wbase + mw) * EPR + E_ + n] = (__bf16)hl[mw * HSTR + n];
    }
}

// ---------------------------------------------------------------- word input GEMM: G = X @ Ww_ih^T + bw  (bf16 WMMA)
__global__ __launch_bounds__(256) void k_gemm_x(
    const __bf16* __restrict__ Xbf, const __bf16* __restrict__ Wihb,
    const float* __restrict__ bw,   float* __restrict__ G)
{
    const int tid  = threadIdx.x, lane = tid & 31, wave = tid >> 5;
    const int tile = blockIdx.x * 8 + wave;          // (S/16)*(4096/16) tiles
    const int mt   = tile >> 8;                      // 256 N-tiles
    const int nt   = tile & 255;
    const int m0   = mt * 16, n0 = nt * 16;
    const int l15  = lane & 15, hi = lane >> 4;
    const int ko   = hi * 8;                         // bf16 fragment K offset
    const __bf16* arow = Xbf  + (size_t)(m0 + l15) * EPR;
    const __bf16* brow = Wihb + (size_t)(n0 + l15) * EPR;

    v8f acc;
    #pragma unroll
    for (int r = 0; r < 8; ++r) acc[r] = 0.0f;

    #pragma unroll 4
    for (int k0 = 0; k0 < EPR; k0 += 32) {
        v8bf alo = *(const v8bf*)(arow + k0 + ko);
        v8bf ahi = *(const v8bf*)(arow + k0 + 16 + ko);
        v8bf blo = *(const v8bf*)(brow + k0 + ko);
        v8bf bhi = *(const v8bf*)(brow + k0 + 16 + ko);
        v16bf a = __builtin_shufflevector(alo, ahi, 0,1,2,3,4,5,6,7,8,9,10,11,12,13,14,15);
        v16bf b = __builtin_shufflevector(blo, bhi, 0,1,2,3,4,5,6,7,8,9,10,11,12,13,14,15);
        acc = __builtin_amdgcn_wmma_f32_16x16x32_bf16(false, a, false, b, (short)0, acc, false, false);
    }
    const int col = n0 + l15;
    const float bias = bw[col];
    #pragma unroll
    for (int r = 0; r < 8; ++r)
        G[(size_t)(m0 + r + hi * 8) * G4H + col] = acc[r] + bias;
}

// ---------------------------------------------------------------- word LSTM recurrence: 64 persistent blocks,
// 16 hidden units per block, spin-barrier global sync per step.
__global__ __launch_bounds__(256) void k_word_lstm(
    const float* __restrict__ G, const __bf16* __restrict__ Whhb,
    float* hbuf, float* __restrict__ hs, unsigned int* cnt)
{
    __shared__ float ldsh[H_];
    __shared__ float red[64][4];
    __shared__ float gat[64];
    __shared__ float cl[16];

    const int tid  = threadIdx.x;
    const int rowL = tid >> 2;          // 0..63 local gate row
    const int q    = tid & 3;           // K quarter (256 each)
    const int g    = rowL >> 4;         // gate {i,f,g,o}
    const int uL   = rowL & 15;         // unit within block
    const int u0   = blockIdx.x * 16;
    const int grow = g * H_ + u0 + uL;  // global gate row
    const v8bf* wrow = (const v8bf*)(Whhb + (size_t)grow * H_ + q * 256);

    if (tid < 16) cl[tid] = 0.0f;

    #pragma unroll 1
    for (int t = 0; t < S_; ++t) {
        for (int k = tid; k < H_; k += 256)
            ldsh[k] = __hip_atomic_load(&hbuf[k], __ATOMIC_RELAXED, __HIP_MEMORY_SCOPE_AGENT);
        __syncthreads();

        float p = 0.0f;
        const float* hq = ldsh + q * 256;
        #pragma unroll 4
        for (int i = 0; i < 32; ++i) {
            v8bf w8 = wrow[i];
            #pragma unroll
            for (int j = 0; j < 8; ++j) p += (float)w8[j] * hq[i * 8 + j];
        }
        red[rowL][q] = p;
        __syncthreads();

        if (tid < 64)
            gat[tid] = G[(size_t)t * G4H + (tid >> 4) * H_ + u0 + (tid & 15)]
                     + red[tid][0] + red[tid][1] + red[tid][2] + red[tid][3];
        __syncthreads();

        if (tid < 16) {
            float iv = gat[tid], fv = gat[16 + tid], gv = gat[32 + tid], ov = gat[48 + tid];
            float cn = sigf(fv) * cl[tid] + sigf(iv) * tanhf(gv);
            float hn = sigf(ov) * tanhf(cn);
            cl[tid] = cn;
            __hip_atomic_store(&hbuf[u0 + tid], hn, __ATOMIC_RELAXED, __HIP_MEMORY_SCOPE_AGENT);
            hs[(size_t)t * H_ + u0 + tid] = hn;
        }
        __threadfence();
        __syncthreads();
        if (tid == 0) {
            __hip_atomic_fetch_add(cnt, 1u, __ATOMIC_RELEASE, __HIP_MEMORY_SCOPE_AGENT);
            const unsigned int tgt = (unsigned int)NBLK_W * (unsigned int)(t + 1);
            while (__hip_atomic_load(cnt, __ATOMIC_ACQUIRE, __HIP_MEMORY_SCOPE_AGENT) < tgt)
                __builtin_amdgcn_s_sleep(8);
        }
        __syncthreads();
    }
}

// ---------------------------------------------------------------- output layer + log_softmax
__global__ __launch_bounds__(64) void k_out(
    const float* __restrict__ hs, const float* __restrict__ Wout,
    const float* __restrict__ bout, float* __restrict__ out)
{
    __shared__ float hrow[H_];
    __shared__ float ev[L_];
    __shared__ float stat[2];
    const int s = blockIdx.x, tid = threadIdx.x;
    for (int k = tid; k < H_; k += 64) hrow[k] = hs[(size_t)s * H_ + k];
    __syncthreads();
    float e = 0.0f;
    if (tid < L_) {
        const float* wr = Wout + (size_t)tid * H_;
        float a = bout[tid];
        #pragma unroll 4
        for (int k = 0; k < H_; ++k) a += wr[k] * hrow[k];
        e = a; ev[tid] = a;
    }
    __syncthreads();
    if (tid == 0) {
        float mx = ev[0];
        for (int j = 1; j < L_; ++j) mx = fmaxf(mx, ev[j]);
        float se = 0.0f;
        for (int j = 0; j < L_; ++j) se += expf(ev[j] - mx);
        stat[0] = mx; stat[1] = logf(se);
    }
    __syncthreads();
    if (tid < L_) out[(size_t)s * L_ + tid] = e - stat[0] - stat[1];
}

// ---------------------------------------------------------------- launcher
extern "C" void kernel_launch(void* const* d_in, const int* in_sizes, int n_in,
                              void* d_out, int out_size, void* d_ws, size_t ws_size,
                              hipStream_t stream)
{
    (void)in_sizes; (void)n_in; (void)out_size; (void)ws_size;
    const int*   sent  = (const int*)  d_in[0];
    const float* chars = (const float*)d_in[1];
    const float* emb   = (const float*)d_in[2];
    const float* Wcih  = (const float*)d_in[3];
    const float* Wchh  = (const float*)d_in[4];
    const float* bc    = (const float*)d_in[5];
    const float* Wwih  = (const float*)d_in[6];
    const float* Wwhh  = (const float*)d_in[7];
    const float* bw    = (const float*)d_in[8];
    const float* Wo    = (const float*)d_in[9];
    const float* bo    = (const float*)d_in[10];
    float* out = (float*)d_out;

    char* p = (char*)d_ws;
    auto carve = [&](size_t bytes) { char* r = p; p += (bytes + 255) & ~(size_t)255; return r; };
    __bf16* Xbf  = (__bf16*)carve((size_t)S_ * EPR * 2);      // 12 MB
    float*  G    = (float*) carve((size_t)S_ * G4H * 4);      // 128 MB
    float*  hs   = (float*) carve((size_t)S_ * H_ * 4);       // 32 MB
    __bf16* Wihb = (__bf16*)carve((size_t)G4H * EPR * 2);     // 6 MB
    __bf16* Whhb = (__bf16*)carve((size_t)G4H * H_ * 2);      // 8 MB
    float*  hbuf = (float*) carve(H_ * 4);
    unsigned int* cnt = (unsigned int*)carve(256);

    k_init<<<1, 1024, 0, stream>>>(hbuf, cnt);
    {
        int n = G4H * EPR;
        k_f32_to_bf16<<<(n + 255) / 256, 256, 0, stream>>>(Wwih, Wihb, n);
    }
    {
        int n = G4H * H_;
        k_f32_to_bf16<<<(n + 255) / 256, 256, 0, stream>>>(Wwhh, Whhb, n);
    }
    k_embed<<<S_, 128, 0, stream>>>(sent, emb, Xbf);
    k_char_lstm<<<S_ / 16, 256, 0, stream>>>(chars, Wcih, Wchh, bc, Xbf);
    k_gemm_x<<<(S_ / 16) * (G4H / 16) / 8, 256, 0, stream>>>(Xbf, Wihb, bw, G);
    k_word_lstm<<<NBLK_W, 256, 0, stream>>>(G, Whhb, hbuf, hs, cnt);
    k_out<<<S_, 64, 0, stream>>>(hs, Wo, bo, out);
}
